// Attention_22797686407397
// MI455X (gfx1250) — compile-verified, matrix operand order checked by reading, not run on hardware
//
#include <hip/hip_runtime.h>
#include <hip/hip_bf16.h>
#include <hip/hip_fp16.h>
#include <cstddef>

typedef __attribute__((ext_vector_type(16))) _Float16 v16h;
typedef __attribute__((ext_vector_type(8)))  _Float16 v8h;
typedef __attribute__((ext_vector_type(8)))  float    v8f;

#define DIMC   1024
#define NHEADS 16
#define HD     64
#define SEQ    2048
#define BATCH  4
#define MTOT   (BATCH * SEQ)   // 8192
#define QSCALE 0.125f          // 64^-0.5

// ---------------------------------------------------------------------------
// Load a 16x32 f16 A-fragment (or a B-fragment of W^T when W is [N,K] row-
// major) from a row-major matrix with leading dimension `ld`.
// CDNA5 16-bit A layout: lane L(0..15) holds row M=L; elements 0..7 are
// K = hi*8..hi*8+7 and elements 8..15 are K = 16+hi*8.. (hi = lane>=16).
// Each lane therefore needs two contiguous 8-half (16B) runs.
// ---------------------------------------------------------------------------
__device__ __forceinline__ v16h frag_ld(const _Float16* __restrict__ p, int ld, int lane) {
    const int r  = lane & 15;
    const int hi = lane >> 4;
    const _Float16* row = p + (size_t)r * ld + hi * 8;
    v8h lo  = *(const v8h*)(row);
    v8h hi8 = *(const v8h*)(row + 16);
    v16h o;
#pragma unroll
    for (int i = 0; i < 8; ++i) { o[i] = lo[i]; o[i + 8] = hi8[i]; }
    return o;
}

__device__ __forceinline__ v8f wmma_f16(v16h a, v16h b, v8f c) {
    return __builtin_amdgcn_wmma_f32_16x16x32_f16(false, a, false, b, (short)0, c, false, false);
}

// ---------------------------------------------------------------------------
// elementwise f32 -> f16 pack
// ---------------------------------------------------------------------------
__global__ void pack_f16_kernel(const float* __restrict__ src, _Float16* __restrict__ dst, int n) {
    int i = blockIdx.x * blockDim.x + threadIdx.x;
    if (i < n) dst[i] = (_Float16)src[i];
}

// ---------------------------------------------------------------------------
// QKV GEMM:  qkv[m, n] = sum_k X[m,k] * Wqkv[n,k]   (M=8192, N=3072, K=1024)
// block tile 128x128, 8 waves (4 in M x 2 in N), wave tile 32x64.
// Epilogue scatters into Q/K/V [B,H,N,hd] f16; Q gets the 1/sqrt(hd) scale.
// ---------------------------------------------------------------------------
__global__ __launch_bounds__(256) void qkv_gemm_kernel(
    const _Float16* __restrict__ X,   // [8192,1024]
    const _Float16* __restrict__ W,   // [3072,1024]
    _Float16* __restrict__ Q, _Float16* __restrict__ K, _Float16* __restrict__ V)
{
    const int lane = threadIdx.x & 31;
    const int wave = threadIdx.x >> 5;
    const int wm   = wave & 3;
    const int wn   = wave >> 2;
    const int m0   = blockIdx.x * 128 + wm * 32;
    const int n0   = blockIdx.y * 128 + wn * 64;

    v8f acc[2][4];
#pragma unroll
    for (int i = 0; i < 2; ++i)
#pragma unroll
        for (int j = 0; j < 4; ++j) acc[i][j] = (v8f){};

    for (int k0 = 0; k0 < DIMC; k0 += 32) {
        __builtin_prefetch(X + (size_t)m0 * DIMC + k0 + 64, 0, 3);
        v16h a0 = frag_ld(X + (size_t)(m0 +  0) * DIMC + k0, DIMC, lane);
        v16h a1 = frag_ld(X + (size_t)(m0 + 16) * DIMC + k0, DIMC, lane);
        v16h b0 = frag_ld(W + (size_t)(n0 +  0) * DIMC + k0, DIMC, lane);
        v16h b1 = frag_ld(W + (size_t)(n0 + 16) * DIMC + k0, DIMC, lane);
        v16h b2 = frag_ld(W + (size_t)(n0 + 32) * DIMC + k0, DIMC, lane);
        v16h b3 = frag_ld(W + (size_t)(n0 + 48) * DIMC + k0, DIMC, lane);
        acc[0][0] = wmma_f16(a0, b0, acc[0][0]);
        acc[0][1] = wmma_f16(a0, b1, acc[0][1]);
        acc[0][2] = wmma_f16(a0, b2, acc[0][2]);
        acc[0][3] = wmma_f16(a0, b3, acc[0][3]);
        acc[1][0] = wmma_f16(a1, b0, acc[1][0]);
        acc[1][1] = wmma_f16(a1, b1, acc[1][1]);
        acc[1][2] = wmma_f16(a1, b2, acc[1][2]);
        acc[1][3] = wmma_f16(a1, b3, acc[1][3]);
    }

    // C layout: lane: col = (lane&15) within 16-tile; row = r + 8*(lane>=16)
    const int cn = lane & 15;
    const int hi = lane >> 4;
#pragma unroll
    for (int i = 0; i < 2; ++i) {
#pragma unroll
        for (int j = 0; j < 4; ++j) {
            const int gn    = n0 + j * 16 + cn;     // 0..3071
            const int which = gn >> 10;             // 0=q 1=k 2=v
            const int c     = gn & (DIMC - 1);
            const int h     = c >> 6;
            const int d     = c & (HD - 1);
            _Float16* dst   = (which == 0) ? Q : ((which == 1) ? K : V);
            const float sc  = (which == 0) ? QSCALE : 1.0f;
#pragma unroll
            for (int r = 0; r < 8; ++r) {
                const int gm   = m0 + i * 16 + r + hi * 8;  // 0..8191
                const int b    = gm >> 11;
                const int nseq = gm & (SEQ - 1);
                const size_t off = (((size_t)(b * NHEADS + h)) * SEQ + nseq) * HD + d;
                dst[off] = (_Float16)(acc[i][j][r] * sc);
            }
        }
    }
}

// ---------------------------------------------------------------------------
// Flash attention: one block per (b,h, 128-query tile); 8 waves, each wave
// owns 16 query rows and a 16x64 f32 O accumulator (4 WMMA C tiles).
// 32-key blocks staged in LDS (K as-is, V transposed), online softmax.
// Q is pre-scaled by 1/sqrt(hd) in the QKV epilogue.
// ---------------------------------------------------------------------------
__global__ __launch_bounds__(256) void flash_attn_kernel(
    const _Float16* __restrict__ Q,
    const _Float16* __restrict__ K,
    const _Float16* __restrict__ V,
    _Float16* __restrict__ O)         // [B, N, C] f16
{
    __shared__ __align__(16) _Float16 sK [32 * HD];      // [key][d]
    __shared__ __align__(16) _Float16 sVt[HD * 32];      // [d][key]
    __shared__ __align__(16) _Float16 sP [8 * 16 * 32];  // per-wave P tile

    const int lane = threadIdx.x & 31;
    const int wave = threadIdx.x >> 5;
    const int bh   = blockIdx.y;                 // 0..63
    const int b    = bh >> 4;
    const int h    = bh & 15;
    const int q0   = blockIdx.x * 128 + wave * 16;

    const _Float16* Qb = Q + (size_t)bh * SEQ * HD;
    const _Float16* Kb = K + (size_t)bh * SEQ * HD;
    const _Float16* Vb = V + (size_t)bh * SEQ * HD;

    // Q tile 16x64 as two A fragments (K-dim = head dim = 64)
    const v16h qf0 = frag_ld(Qb + (size_t)q0 * HD +  0, HD, lane);
    const v16h qf1 = frag_ld(Qb + (size_t)q0 * HD + 32, HD, lane);

    v8f oacc[4];
#pragma unroll
    for (int j = 0; j < 4; ++j) oacc[j] = (v8f){};
    float mrun[8], lrun[8];
#pragma unroll
    for (int r = 0; r < 8; ++r) { mrun[r] = -3.0e38f; lrun[r] = 0.f; }

    for (int kb = 0; kb < SEQ; kb += 32) {
        __syncthreads();  // previous iteration finished consuming LDS
        {   // cooperative stage: K block (2048 halves, 16B per thread)
            const int t = threadIdx.x;
            ((v8h*)sK)[t] = ((const v8h*)(Kb + (size_t)kb * HD))[t];
            // V block, transposed into sVt[d][key]
            const int key = t & 31, dg = t >> 5;
            v8h vv = *(const v8h*)(Vb + (size_t)(kb + key) * HD + dg * 8);
#pragma unroll
            for (int e = 0; e < 8; ++e) sVt[(dg * 8 + e) * 32 + key] = vv[e];
        }
        __syncthreads();

        // S = Q @ K^T for two 16-key tiles (chain two k=32 WMMAs each)
        v8f s[2];
#pragma unroll
        for (int t = 0; t < 2; ++t) {
            v16h kf0 = frag_ld(sK + (t * 16) * HD +  0, HD, lane);
            v16h kf1 = frag_ld(sK + (t * 16) * HD + 32, HD, lane);
            v8f c = (v8f){};
            c = wmma_f16(qf0, kf0, c);
            c = wmma_f16(qf1, kf1, c);
            s[t] = c;
        }

        // online softmax; row m = r + 8*hi lives across 16 lanes (same half)
        float p0[8], p1[8];
#pragma unroll
        for (int r = 0; r < 8; ++r) {
            float mb = fmaxf(s[0][r], s[1][r]);
            mb = fmaxf(mb, __shfl_xor(mb, 1));
            mb = fmaxf(mb, __shfl_xor(mb, 2));
            mb = fmaxf(mb, __shfl_xor(mb, 4));
            mb = fmaxf(mb, __shfl_xor(mb, 8));
            const float mnew = fmaxf(mrun[r], mb);
            const float fac  = __expf(mrun[r] - mnew);
            const float e0   = __expf(s[0][r] - mnew);
            const float e1   = __expf(s[1][r] - mnew);
            float rs = e0 + e1;
            rs += __shfl_xor(rs, 1);
            rs += __shfl_xor(rs, 2);
            rs += __shfl_xor(rs, 4);
            rs += __shfl_xor(rs, 8);
            lrun[r] = lrun[r] * fac + rs;
            mrun[r] = mnew;
#pragma unroll
            for (int j = 0; j < 4; ++j) oacc[j][r] *= fac;
            p0[r] = e0; p1[r] = e1;
        }

        // C-layout -> A-layout for P via per-wave LDS bounce, then P @ V
        {
            _Float16* pp = sP + wave * (16 * 32);
            const int cn = lane & 15, hh = lane >> 4;
#pragma unroll
            for (int r = 0; r < 8; ++r) {
                pp[(r + hh * 8) * 32 + cn]      = (_Float16)p0[r];
                pp[(r + hh * 8) * 32 + 16 + cn] = (_Float16)p1[r];
            }
            const v16h pf = frag_ld(pp, 32, lane);
#pragma unroll
            for (int j = 0; j < 4; ++j) {
                v16h vf = frag_ld(sVt + (j * 16) * 32, 32, lane);
                oacc[j] = wmma_f16(pf, vf, oacc[j]);
            }
        }
    }

    // normalize and store to [B, N, C] f16
    const int cn = lane & 15, hh = lane >> 4;
#pragma unroll
    for (int r = 0; r < 8; ++r) {
        const float inv = 1.0f / lrun[r];
        const int gq = q0 + r + hh * 8;
        const size_t base = ((size_t)b * SEQ + gq) * DIMC + h * HD;
#pragma unroll
        for (int j = 0; j < 4; ++j)
            O[base + j * 16 + cn] = (_Float16)(oacc[j][r] * inv);
    }
}

// ---------------------------------------------------------------------------
// Projection GEMM: out[m,n] = sum_k A[m,k] * Wp[n,k] + bias[n]  (f32 out)
// ---------------------------------------------------------------------------
__global__ __launch_bounds__(256) void proj_gemm_kernel(
    const _Float16* __restrict__ A,    // [8192,1024]
    const _Float16* __restrict__ W,    // [1024,1024]
    const float* __restrict__ bias,    // [1024]
    float* __restrict__ out)           // [8192,1024]
{
    const int lane = threadIdx.x & 31;
    const int wave = threadIdx.x >> 5;
    const int wm   = wave & 3;
    const int wn   = wave >> 2;
    const int m0   = blockIdx.x * 128 + wm * 32;
    const int n0   = blockIdx.y * 128 + wn * 64;

    v8f acc[2][4];
#pragma unroll
    for (int i = 0; i < 2; ++i)
#pragma unroll
        for (int j = 0; j < 4; ++j) acc[i][j] = (v8f){};

    for (int k0 = 0; k0 < DIMC; k0 += 32) {
        __builtin_prefetch(A + (size_t)m0 * DIMC + k0 + 64, 0, 3);
        v16h a0 = frag_ld(A + (size_t)(m0 +  0) * DIMC + k0, DIMC, lane);
        v16h a1 = frag_ld(A + (size_t)(m0 + 16) * DIMC + k0, DIMC, lane);
        v16h b0 = frag_ld(W + (size_t)(n0 +  0) * DIMC + k0, DIMC, lane);
        v16h b1 = frag_ld(W + (size_t)(n0 + 16) * DIMC + k0, DIMC, lane);
        v16h b2 = frag_ld(W + (size_t)(n0 + 32) * DIMC + k0, DIMC, lane);
        v16h b3 = frag_ld(W + (size_t)(n0 + 48) * DIMC + k0, DIMC, lane);
        acc[0][0] = wmma_f16(a0, b0, acc[0][0]);
        acc[0][1] = wmma_f16(a0, b1, acc[0][1]);
        acc[0][2] = wmma_f16(a0, b2, acc[0][2]);
        acc[0][3] = wmma_f16(a0, b3, acc[0][3]);
        acc[1][0] = wmma_f16(a1, b0, acc[1][0]);
        acc[1][1] = wmma_f16(a1, b1, acc[1][1]);
        acc[1][2] = wmma_f16(a1, b2, acc[1][2]);
        acc[1][3] = wmma_f16(a1, b3, acc[1][3]);
    }

    const int cn = lane & 15;
    const int hi = lane >> 4;
#pragma unroll
    for (int i = 0; i < 2; ++i) {
#pragma unroll
        for (int j = 0; j < 4; ++j) {
            const int gn = n0 + j * 16 + cn;
            const float bv = bias[gn];
#pragma unroll
            for (int r = 0; r < 8; ++r) {
                const int gm = m0 + i * 16 + r + hi * 8;
                out[(size_t)gm * DIMC + gn] = acc[i][j][r] + bv;
            }
        }
    }
}

// ---------------------------------------------------------------------------
// Host launcher. Workspace layout (f16, all MB-aligned; total 88 MB):
//   [ 0MB) xh    8192x1024
//   [16MB) wqkvh 3072x1024
//   [22MB) wprojh 1024x1024
//   [24MB) Q  [B,H,N,hd]   (pre-scaled)
//   [40MB) K  [B,H,N,hd]
//   [56MB) V  [B,H,N,hd]
//   [72MB) attn out [B,N,C]
// ---------------------------------------------------------------------------
extern "C" void kernel_launch(void* const* d_in, const int* in_sizes, int n_in,
                              void* d_out, int out_size, void* d_ws, size_t ws_size,
                              hipStream_t stream) {
    const float* x      = (const float*)d_in[0];
    const float* w_qkv  = (const float*)d_in[1];
    const float* w_proj = (const float*)d_in[2];
    const float* b_proj = (const float*)d_in[3];
    float* out = (float*)d_out;

    char* ws = (char*)d_ws;
    _Float16* xh  = (_Float16*)(ws + ((size_t) 0 << 20));
    _Float16* wqh = (_Float16*)(ws + ((size_t)16 << 20));
    _Float16* wph = (_Float16*)(ws + ((size_t)22 << 20));
    _Float16* q   = (_Float16*)(ws + ((size_t)24 << 20));
    _Float16* k   = (_Float16*)(ws + ((size_t)40 << 20));
    _Float16* v   = (_Float16*)(ws + ((size_t)56 << 20));
    _Float16* ao  = (_Float16*)(ws + ((size_t)72 << 20));

    const int nx = MTOT * DIMC;       // 8388608
    const int nq = 3 * DIMC * DIMC;   // 3145728
    const int np = DIMC * DIMC;       // 1048576
    pack_f16_kernel<<<nx / 256, 256, 0, stream>>>(x, xh, nx);
    pack_f16_kernel<<<nq / 256, 256, 0, stream>>>(w_qkv, wqh, nq);
    pack_f16_kernel<<<np / 256, 256, 0, stream>>>(w_proj, wph, np);

    qkv_gemm_kernel<<<dim3(MTOT / 128, 3 * DIMC / 128), 256, 0, stream>>>(xh, wqh, q, k, v);
    flash_attn_kernel<<<dim3(SEQ / 128, BATCH * NHEADS), 256, 0, stream>>>(q, k, v, ao);
    proj_gemm_kernel<<<dim3(MTOT / 128, DIMC / 128), 256, 0, stream>>>(ao, wph, b_proj, out);
}
